// C3_2164663517637
// MI455X (gfx1250) — matrix operand order repaired and checked by reading, not compiled, not run on hardware
//
#include <hip/hip_runtime.h>

typedef __attribute__((ext_vector_type(16))) _Float16 v16h;
typedef __attribute__((ext_vector_type(8)))  float    v8f;
typedef __attribute__((ext_vector_type(4)))  unsigned int u32x4;
typedef __attribute__((ext_vector_type(4)))  int  i32x4;
typedef __attribute__((ext_vector_type(8)))  int  i32x8;

#define IN_C 6
#define OUT_C 16
#define H 512
#define W 512
#define OH 508
#define OW 508
#define TY 16
#define TX 64
#define LH (TY + 4)    // 20 rows incl. halo
#define LW (TX + 4)    // 68 cols incl. halo
#define KREAL 150      // 6 * 25
#define KDIM  160      // padded to 5 * 32

// channel maps for the grouped convs (flattened)
__device__ __constant__ int d_MAPS3[18] = {
  0,1,2, 1,2,3, 2,3,4, 3,4,5, 0,4,5, 0,1,5 };
__device__ __constant__ int d_MAPS4[36] = {
  0,1,2,3, 1,2,3,4, 2,3,4,5, 0,3,4,5, 0,1,4,5,
  0,1,2,5, 0,1,3,4, 1,2,4,5, 0,2,3,5 };
__device__ __constant__ int d_MAP6[6] = { 0,1,2,3,4,5 };

// compile-time im2col offset (in elements) for K index kp = c*25 + ky*5 + kx
__host__ __device__ constexpr int koff_c(int kp) {
  int kk = (kp < KREAL) ? kp : 0;
  int c = kk / 25, r = kk % 25;
  return (c * LH + r / 5) * LW + (r % 5);
}

// ---------------------------------------------------------------------------
// Kernel 1: binarize + scatter into dense padded f16 weight matrix Wd[16][160]
// ---------------------------------------------------------------------------
__global__ void prep_weights(const float* __restrict__ w3,
                             const float* __restrict__ w4,
                             const float* __restrict__ w6,
                             _Float16* __restrict__ Wd) {
  int o = threadIdx.x;
  if (o >= OUT_C) return;
  for (int k = 0; k < KDIM; ++k) Wd[o * KDIM + k] = (_Float16)0.f;

  const float* w; int nc; const int* map;
  if (o < 6)       { w = w3 + o * 3 * 25;       nc = 3; map = &d_MAPS3[o * 3]; }
  else if (o < 15) { w = w4 + (o - 6) * 4 * 25; nc = 4; map = &d_MAPS4[(o - 6) * 4]; }
  else             { w = w6;                    nc = 6; map = d_MAP6; }

  float c[6 * 25];
  for (int p = 0; p < 25; ++p) {
    float mean = 0.f;
    for (int ci = 0; ci < nc; ++ci) mean += w[ci * 25 + p];
    mean /= (float)nc;
    for (int ci = 0; ci < nc; ++ci) {
      float v = w[ci * 25 + p] - mean;
      c[ci * 25 + p] = fminf(1.f, fmaxf(-1.f, v));
    }
  }
  float alpha = 0.f;
  for (int i = 0; i < nc * 25; ++i) alpha += fabsf(c[i]);
  alpha /= (float)(nc * 25);
  for (int ci = 0; ci < nc; ++ci)
    for (int p = 0; p < 25; ++p) {
      float v = c[ci * 25 + p];
      float s = (v > 0.f) ? alpha : ((v < 0.f) ? -alpha : 0.f);
      Wd[o * KDIM + map[ci] * 25 + p] = (_Float16)s;
    }
}

// ---------------------------------------------------------------------------
// Kernel 2: implicit-GEMM 5x5 conv, 6ch -> 16ch, via v_wmma_f32_16x16x32_f16
// Input tile staged by the Tensor Data Mover (tensor_load_to_lds).
// block = 256 threads (8 waves): 4 waves across x (16 cols), 2 across y (8 rows)
// block output tile: 16 ch x 16 rows x 64 cols
// ---------------------------------------------------------------------------
__global__ __launch_bounds__(256)
void conv_wmma(const float* __restrict__ x,
               const _Float16* __restrict__ Wd,
               float* __restrict__ out) {
  __shared__ float    ldsx[IN_C * LH * LW];   // 32640 B fp32 input tile + halo
  __shared__ _Float16 ldsw[OUT_C * KDIM];     // 5120 B dense weights

  const int b  = blockIdx.z;
  const int y0 = blockIdx.y * TY;
  const int x0 = blockIdx.x * TX;
  const int tid = threadIdx.x;

  // ---- stage weights to LDS (b32 copies) ----
  {
    const unsigned int* wsrc = (const unsigned int*)Wd;
    unsigned int* wdst = (unsigned int*)ldsw;
    for (int i = tid; i < (OUT_C * KDIM) / 2; i += 256) wdst[i] = wsrc[i];
  }

#if defined(__has_builtin) && __has_builtin(__builtin_amdgcn_tensor_load_to_lds)
  // ---- TDM: one DMA of the 68x20x6 fp32 tile, zero-fill past tensor edge ----
  if (tid < 32) {
    unsigned long long ga =
        (unsigned long long)(const void*)(x + ((size_t)(b * IN_C) * H + y0) * W + x0);
    unsigned lds_base = (unsigned)(unsigned long long)(void*)ldsx; // low 32b = LDS offset
    unsigned td0 = (unsigned)(W - x0);   // valid cols from tile start (OOB -> 0)
    unsigned td1 = (unsigned)(H - y0);   // valid rows from tile start (OOB -> 0)
    const unsigned s0 = (unsigned)W;         // row stride (elements)
    const unsigned s1 = (unsigned)(W * H);   // channel stride (elements)

    u32x4 g0 = { 1u,                                   // count=1, user descriptor
                 lds_base,                              // lds_addr
                 (unsigned)(ga & 0xffffffffu),          // global_addr[31:0]
                 ((unsigned)((ga >> 32) & 0x1ffffffu)) | (2u << 30) }; // [56:32]+type=2
    i32x8 g1 = { (int)(2u << 16),                       // data_size=4B, no extras
                 (int)((td0 & 0xffffu) << 16),          // tensor_dim0[15:0]
                 (int)((td0 >> 16) | ((td1 & 0xffffu) << 16)),
                 (int)((td1 >> 16) | ((unsigned)LW << 16)),   // tile_dim0=68
                 (int)((unsigned)LH | ((unsigned)IN_C << 16)),// tile_dim1=20, tile_dim2=6
                 (int)s0,                               // tensor_dim0_stride[31:0]
                 (int)((s0 >> 16 >> 16) | ((s1 & 0xffffu) << 16)), // d0s[47:32] | d1s[15:0]
                 (int)(s1 >> 16) };                     // tensor_dim1_stride[47:16]
    i32x4 g2 = { IN_C, 1, 0, 0 };                       // tensor_dim2=6
    i32x4 g3 = { 0, 0, 0, 0 };
    i32x8 g4 = { 0, 0, 0, 0, 0, 0, 0, 0 };              // unused extra group
    __builtin_amdgcn_tensor_load_to_lds(g0, g1, g2, g3, g4, 0);
  }
  __builtin_amdgcn_s_wait_tensorcnt(0);
#else
  // ---- fallback: manual staging fp32 global -> LDS ----
  {
    const int total = IN_C * LH * LW;
    for (int idx = tid; idx < total; idx += 256) {
      int c   = idx / (LH * LW);
      int rem = idx % (LH * LW);
      int r   = rem / LW, col = rem % LW;
      int gy = y0 + r, gx = x0 + col;
      float v = 0.f;
      if (gy < H && gx < W) v = x[((b * IN_C + c) * H + gy) * W + gx];
      ldsx[idx] = v;
    }
  }
#endif
  // speculative prefetch of the neighboring x-tile (global_prefetch_b8)
  {
    int gy = y0 + (tid >> 3), gx = x0 + TX + ((tid & 7) << 3);
    if (gy < H && gx < W) __builtin_prefetch(&x[((b * IN_C) * H + gy) * W + gx], 0, 1);
  }
  __syncthreads();

  const int lane = tid & 31;
  const int wave = tid >> 5;
  const int wx   = wave & 3;        // x sub-tile: cols wx*16 .. wx*16+15
  const int wy   = wave >> 2;       // y sub-tile: rows wy*8 .. wy*8+7
  const int half = lane >> 4;       // upper/lower half-wave
  const int l16  = lane & 15;

  // ---- A fragments (weights), CDNA5 16-bit A layout ----
  // lane holds M = lane&15; element i -> K = (i<8 ? i : i+8) + 8*half
  v16h a[5];
#pragma unroll
  for (int s = 0; s < 5; ++s) {
#pragma unroll
    for (int i = 0; i < 16; ++i) {
      int k = 32 * s + (i < 8 ? i : i + 8) + half * 8;
      a[s][i] = ldsw[l16 * KDIM + k];
    }
  }

#pragma unroll 1
  for (int row = 0; row < 8; ++row) {
    const int yy = wy * 8 + row;                       // row within tile
    const int rowbase = yy * LW + wx * 16 + l16;       // per-lane LDS base (elements)
    v8f acc = {};

#pragma unroll
    for (int s = 0; s < 5; ++s) {
      // B fragment, CDNA5 16-bit B layout:
      // lane holds N = lane&15; element i -> K = (i<8 ? i : i+8) + 8*half
      v16h bf;
#pragma unroll
      for (int i = 0; i < 16; ++i) {
        const int k0 = 32 * s + (i < 8 ? i : i + 8);      // half == 0
        const int k1 = 32 * s + (i < 8 ? i : i + 8) + 8;  // half == 1
        const int off = half ? koff_c(k1) : koff_c(k0);   // two constants, cndmask
        float fv = ldsx[rowbase + off];
        _Float16 v = (_Float16)fv;
        if (k1 >= KREAL) {                // compile-time pruned except s==4 tail
          if (half) v = (_Float16)0.f;
        }
        if (k0 >= KREAL) {
          if (!half) v = (_Float16)0.f;
        }
        bf[i] = v;
      }
      acc = __builtin_amdgcn_wmma_f32_16x16x32_f16(
          false, a[s], false, bf, (short)0, acc, false, false);
    }

    // ---- store: C/D layout -> lane gives N (out col), VGPR r gives M (ch) ----
    const int gy = y0 + yy;
    const int gx = x0 + wx * 16 + l16;
    if (gy < OH && gx < OW) {
#pragma unroll
      for (int r8 = 0; r8 < 8; ++r8) {
        const int ch = r8 + 8 * half;
        out[((b * OUT_C + ch) * OH + gy) * OW + gx] = acc[r8];
      }
    }
  }
}

extern "C" void kernel_launch(void* const* d_in, const int* in_sizes, int n_in,
                              void* d_out, int out_size, void* d_ws, size_t ws_size,
                              hipStream_t stream) {
  const float* x  = (const float*)d_in[0];
  const float* w3 = (const float*)d_in[1];
  const float* w4 = (const float*)d_in[2];
  const float* w6 = (const float*)d_in[3];
  _Float16* Wd = (_Float16*)d_ws;            // needs 16*160*2 = 5120 B
  float* out = (float*)d_out;

  prep_weights<<<1, 32, 0, stream>>>(w3, w4, w6, Wd);

  dim3 grid((OW + TX - 1) / TX,   // 8
            (OH + TY - 1) / TY,   // 32
            32);                  // batch
  conv_wmma<<<grid, 256, 0, stream>>>(x, Wd, out);
}
